// TaDCGNoTime_3332894621732
// MI455X (gfx1250) — matrix operand order repaired and checked by reading, not compiled, not run on hardware
//
#include <hip/hip_runtime.h>

// ---------------------------------------------------------------------------
// MI455X / gfx1250 implementation of the TaDCG "NoTime" forward pass.
// All dense GEMMs run on v_wmma_f32_16x16x32_f16 (wave32, 16x16 tiles,
// f32 accumulate).  The adjacency GEMM (the only one with heavy operand
// reuse) stages its shared B panel in LDS via the Tensor Data Mover
// (tensor_load_to_lds + s_wait_tensorcnt, double buffered) so 8 waves per
// block read it from DS instead of 8x from L2.  Attention stages
// unnormalized exp() tiles through LDS to convert the WMMA C layout into an
// A fragment for the probs @ V^T product.
// ---------------------------------------------------------------------------

typedef __attribute__((ext_vector_type(16))) _Float16 v16h;
typedef __attribute__((ext_vector_type(8)))  _Float16 v8h;
typedef __attribute__((ext_vector_type(8)))  float    v8f;
typedef __attribute__((ext_vector_type(4)))  unsigned int v4u;
typedef __attribute__((ext_vector_type(8)))  int v8i;
typedef __attribute__((ext_vector_type(4)))  int v4i;

#if defined(__has_builtin)
#  if __has_builtin(__builtin_amdgcn_tensor_load_to_lds)
#    define HAVE_TDM 1
#  endif
#endif
#ifndef HAVE_TDM
#  define HAVE_TDM 0
#endif

namespace {
constexpr int kB = 8, kT = 16, kN = 1024, kCS = 64, kG = 64, kH = 128, kA = 32, kOUT = 1024;
constexpr float kInvSqrtA = 0.17677669529663687f;   // 1/sqrt(32)
}

__device__ __forceinline__ v8f zero8() { v8f z = {}; return z; }

__device__ __forceinline__ v16h frag2(const _Float16* p0, const _Float16* p1) {
  v8h a = *(const v8h*)p0;
  v8h b = *(const v8h*)p1;
  v16h r;
#pragma unroll
  for (int i = 0; i < 8; ++i) { r[i] = a[i]; r[i + 8] = b[i]; }
  return r;
}

__device__ __forceinline__ v8f wmma_f16(v16h a, v16h b, v8f c) {
  // D = A(16x32 f16) * B(32x16 f16) + C(16x16 f32)
  return __builtin_amdgcn_wmma_f32_16x16x32_f16(false, a, false, b, (short)0, c, false, false);
}

__device__ __forceinline__ float sigmoidf_(float x) { return 1.0f / (1.0f + __expf(-x)); }

#if HAVE_TDM
// 2-D f16 tile load via the Tensor Data Mover.  D# packed per CDNA5 ISA 8.3/8.4:
// group0: count=1 | lds_addr[63:32] | global_addr[120:64] | type=2 ("image")
// group1: data_size=1 (2B) | tensor_dim0/1 | tile_dim0/1 | tensor_dim0_stride
__device__ __forceinline__ void tdm_load_2d_f16(unsigned lds_addr, unsigned long long gaddr,
                                                int tensor_d0, int tensor_d1,
                                                int tile_d0, int tile_d1, int d0_stride) {
  v4u g0;
  g0[0] = 1u;                                           // count=1, user descriptor
  g0[1] = lds_addr;                                     // lds_addr (bytes)
  g0[2] = (unsigned)(gaddr & 0xffffffffull);            // global_addr[31:0]
  g0[3] = (unsigned)((gaddr >> 32) & 0x1ffffffull) | (2u << 30);  // addr[56:32], type=2
  v8i g1;
  g1[0] = 0x00010000;                                   // wg_mask=0, data_size=1 (2B)
  g1[1] = (tensor_d0 & 0xffff) << 16;                   // tensor_dim0[15:0]
  g1[2] = ((unsigned)tensor_d0 >> 16) | ((unsigned)(tensor_d1 & 0xffff) << 16);
  g1[3] = ((unsigned)tensor_d1 >> 16) | ((unsigned)tile_d0 << 16);   // tile_dim0
  g1[4] = tile_d1 & 0xffff;                             // tile_dim1, tile_dim2=0
  g1[5] = d0_stride;                                    // tensor_dim0_stride[31:0]
  g1[6] = 0;
  g1[7] = 0;
  v4i g2 = {0, 0, 0, 0};
  v4i g3 = {0, 0, 0, 0};
#if __clang_major__ >= 23
  v8i g4 = {0, 0, 0, 0, 0, 0, 0, 0};
  __builtin_amdgcn_tensor_load_to_lds(g0, g1, g2, g3, g4, 0);
#else
  __builtin_amdgcn_tensor_load_to_lds(g0, g1, g2, g3, 0);
#endif
}
#endif  // HAVE_TDM

// ---------------------------------------------------------------------------
// Adjacency GEMM: agg[b] = adj(1024x1024 f16) @ X[b](1024x64).
// Block = 8 waves = 128 rows x 64 cols.  The shared B panel (64 rows of X^T x
// 128 K) is DMA'd into LDS by the TDM (double buffered) and consumed by all
// 8 waves; A rows stream from global/L2 with prefetch.
// ---------------------------------------------------------------------------
__global__ __launch_bounds__(256) void adj_gemm_tdm_kernel(
    const _Float16* __restrict__ A,    // adj_h [1024,1024]
    const _Float16* __restrict__ Xt,   // [B,64,1024] (K-major X)
    float* __restrict__ Cagg) {        // [B,1024,64]
  __shared__ _Float16 bpanel[2][kCS * 128];   // 2 x 16 KB
  const int wid = threadIdx.x >> 5;
  const int lane = threadIdx.x & 31;
  const int lm = lane & 15, half = lane >> 4;
  const int rb = blockIdx.x % (kN / 128);
  const int bi = blockIdx.x / (kN / 128);
  const int rowTile = rb * 8 + wid;
  const _Float16* Ab = A + (long)(rowTile * 16 + lm) * kN;
  const _Float16* Xb = Xt + (long)bi * kCS * kN;

  v8f acc[4];
#pragma unroll
  for (int ct = 0; ct < 4; ++ct) acc[ct] = zero8();

#if HAVE_TDM
  if (wid == 0) {
    tdm_load_2d_f16((unsigned)(unsigned long long)(const void*)&bpanel[0][0],
                    (unsigned long long)(const void*)Xb, kN, kCS, 128, kCS, kN);
    __builtin_amdgcn_s_wait_tensorcnt(0);
  }
  __syncthreads();
  for (int c = 0; c < kN / 128; ++c) {
    const int k0 = c * 128;
    if (c + 1 < kN / 128 && wid == 0) {   // DMA next panel while computing this one
      tdm_load_2d_f16((unsigned)(unsigned long long)(const void*)&bpanel[(c + 1) & 1][0],
                      (unsigned long long)(const void*)(Xb + k0 + 128), kN, kCS, 128, kCS, kN);
    }
    const _Float16* bp = &bpanel[c & 1][0];
    __builtin_prefetch(Ab + k0 + 256, 0, 1);
#pragma unroll
    for (int kk = 0; kk < 4; ++kk) {
      v16h af = frag2(Ab + k0 + kk * 32 + half * 16, Ab + k0 + kk * 32 + half * 16 + 8);
#pragma unroll
      for (int ct = 0; ct < 4; ++ct) {
        const _Float16* brow = bp + (ct * 16 + lm) * 128 + kk * 32 + half * 16;
        acc[ct] = wmma_f16(af, frag2(brow, brow + 8), acc[ct]);
      }
    }
    if (wid == 0) __builtin_amdgcn_s_wait_tensorcnt(0);
    __syncthreads();
  }
#else
  // fallback: cooperative LDS staging (single buffer)
  for (int c = 0; c < kN / 128; ++c) {
    const int k0 = c * 128;
    __syncthreads();
    for (int id = threadIdx.x; id < kCS * 16; id += 256) {
      int row = id >> 4, c8 = id & 15;
      *(v8h*)&bpanel[0][row * 128 + c8 * 8] = *(const v8h*)&Xb[(long)row * kN + k0 + c8 * 8];
    }
    __syncthreads();
#pragma unroll
    for (int kk = 0; kk < 4; ++kk) {
      v16h af = frag2(Ab + k0 + kk * 32 + half * 16, Ab + k0 + kk * 32 + half * 16 + 8);
#pragma unroll
      for (int ct = 0; ct < 4; ++ct) {
        const _Float16* brow = &bpanel[0][(ct * 16 + lm) * 128 + kk * 32 + half * 16];
        acc[ct] = wmma_f16(af, frag2(brow, brow + 8), acc[ct]);
      }
    }
  }
#endif

#pragma unroll
  for (int ct = 0; ct < 4; ++ct) {
    const int col = ct * 16 + lm;
#pragma unroll
    for (int r = 0; r < 8; ++r) {
      const int row = rowTile * 16 + r + 8 * half;
      Cagg[(long)bi * kN * kCS + (long)row * kCS + col] = acc[ct][r];
    }
  }
}

// ---------------------------------------------------------------------------
// Generic WMMA GEMM: C[M,Ncols] = A[M,K] (row-major f16) x Bt[Ncols,K] (K-major
// f16) + bias, optional leaky-relu, f32 / f16 / transposed-f16 outputs.
// One wave per 16x16 output tile.
// ---------------------------------------------------------------------------
__global__ __launch_bounds__(256) void wmma_gemm_kernel(
    const _Float16* __restrict__ A, int lda, long sA,
    const _Float16* __restrict__ Bt, int ldb, long sBt,
    float* __restrict__ C, int ldc, long sC,
    _Float16* __restrict__ Ch,
    _Float16* __restrict__ Ct, int ldct, long sCt,
    const float* __restrict__ bias, int act,
    int K, int tilesM, int tilesN, int totalWaves) {
  int w = blockIdx.x * 8 + (threadIdx.x >> 5);
  if (w >= totalWaves) return;
  int lane = threadIdx.x & 31;
  int lm = lane & 15, half = lane >> 4;
  int nt = w % tilesN;
  int mt = (w / tilesN) % tilesM;
  int bi = w / (tilesN * tilesM);

  const _Float16* Ab = A + (long)bi * sA + (long)(mt * 16 + lm) * lda;
  const _Float16* Bb = Bt + (long)bi * sBt + (long)(nt * 16 + lm) * ldb + half * 16;

  v8f acc = zero8();
  for (int k0 = 0; k0 < K; k0 += 32) {
    __builtin_prefetch(Ab + k0 + 128, 0, 1);   // global_prefetch_b8 next K panel
    v16h af = frag2(Ab + k0 + half * 16, Ab + k0 + half * 16 + 8);
    v16h bf = frag2(Bb + k0, Bb + k0 + 8);
    acc = wmma_f16(af, bf, acc);
  }

  int col = nt * 16 + lm;
  float bv = bias ? bias[col] : 0.0f;
#pragma unroll
  for (int r = 0; r < 8; ++r) {
    int row = mt * 16 + r + 8 * half;
    float v = acc[r] + bv;
    if (act == 1) v = (v > 0.0f) ? v : 0.01f * v;   // jax.nn.leaky_relu default slope
    if (C)  C[(long)bi * sC + (long)row * ldc + col] = v;
    if (Ch) Ch[(long)bi * sC + (long)row * ldc + col] = (_Float16)v;
    if (Ct) Ct[(long)bi * sCt + (long)col * ldct + row] = (_Float16)v;
  }
}

// ---------------------------------------------------------------------------
// Attention: per (batch, 16-row tile) wave.  logits tile = Q(16x32) K^T(32x16)
// via WMMA; masked exp accumulated (no max-subtract: logits are bounded);
// unnormalized probs staged in LDS -> A fragment -> probs @ V^T via WMMA.
// attn = tanh(acc / rowsum), 0 where the key set is empty.
// ---------------------------------------------------------------------------
__global__ __launch_bounds__(256) void attn_kernel(
    const _Float16* __restrict__ Qh,   // [B,N,32]
    const _Float16* __restrict__ Kh,   // [B,N,32]
    const _Float16* __restrict__ Vt,   // [B,128,1024]  (K-major V, bias included)
    const float* __restrict__ m23w,    // [B,N]
    float* __restrict__ attn) {        // [B,N,128]
  __shared__ _Float16 pbuf[8][16 * 32];
  int wslot = threadIdx.x >> 5;
  int w = blockIdx.x * 8 + wslot;          // grid == B*N/16/8 exactly
  int mt = w % (kN / 16);
  int bi = w / (kN / 16);
  int lane = threadIdx.x & 31;
  int lm = lane & 15, half = lane >> 4;

  const _Float16* qrow = Qh + ((long)bi * kN + mt * 16 + lm) * kA;
  v16h qf = frag2(qrow + half * 16, qrow + half * 16 + 8);

  v8f vacc[8];
#pragma unroll
  for (int c = 0; c < 8; ++c) vacc[c] = zero8();
  float rowsum[8];
#pragma unroll
  for (int r = 0; r < 8; ++r) rowsum[r] = 0.0f;

  _Float16* pl = &pbuf[wslot][0];

  for (int n0 = 0; n0 < kN; n0 += 32) {
#pragma unroll
    for (int s = 0; s < 2; ++s) {
      int nb = n0 + s * 16;
      const _Float16* krow = Kh + ((long)bi * kN + nb + lm) * kA + half * 16;
      v16h kf = frag2(krow, krow + 8);
      v8f lt = wmma_f16(qf, kf, zero8());
      float mv = m23w[bi * kN + nb + lm];
#pragma unroll
      for (int r = 0; r < 8; ++r) {
        float e = (mv > 0.0f) ? __expf(lt[r] * kInvSqrtA) : 0.0f;
        rowsum[r] += e;
        pl[(r + 8 * half) * 32 + s * 16 + lm] = (_Float16)e;   // DS in-order per wave
      }
    }
    const _Float16* prow = pl + lm * 32;
    v16h pf = frag2(prow + half * 16, prow + half * 16 + 8);
#pragma unroll
    for (int c = 0; c < 8; ++c) {
      const _Float16* vrow = Vt + (long)bi * kH * kN + (long)(c * 16 + lm) * kN + n0 + half * 16;
      v16h vf = frag2(vrow, vrow + 8);
      vacc[c] = wmma_f16(pf, vf, vacc[c]);
    }
  }

  // row-sum: reduce the 16 column-lanes of each half-group (wave32, width 16)
#pragma unroll
  for (int r = 0; r < 8; ++r) {
    float s = rowsum[r];
    s += __shfl_xor(s, 1, 16);
    s += __shfl_xor(s, 2, 16);
    s += __shfl_xor(s, 4, 16);
    s += __shfl_xor(s, 8, 16);
    rowsum[r] = s;
  }

#pragma unroll
  for (int c = 0; c < 8; ++c) {
#pragma unroll
    for (int r = 0; r < 8; ++r) {
      int row = mt * 16 + r + 8 * half;
      float d = rowsum[r];
      float v = (d > 0.0f) ? tanhf(vacc[c][r] / d) : 0.0f;
      attn[((long)bi * kN + row) * kH + c * 16 + lm] = v;
    }
  }
}

// --------------------------- element-wise kernels --------------------------

__global__ void cvt_f16_kernel(_Float16* dst, const float* src, long n) {
  for (long i = blockIdx.x * (long)blockDim.x + threadIdx.x; i < n;
       i += (long)gridDim.x * blockDim.x)
    dst[i] = (_Float16)src[i];
}

__global__ void transpose_f16_kernel(_Float16* dst, const float* src, int rows, int cols) {
  long n = (long)rows * cols;
  for (long i = blockIdx.x * (long)blockDim.x + threadIdx.x; i < n;
       i += (long)gridDim.x * blockDim.x) {
    int r = (int)(i / cols), c = (int)(i % cols);
    dst[(long)c * rows + r] = (_Float16)src[i];
  }
}

__global__ void zero_f32_kernel(float* p, long n) {
  for (long i = blockIdx.x * (long)blockDim.x + threadIdx.x; i < n;
       i += (long)gridDim.x * blockDim.x)
    p[i] = 0.0f;
}

__global__ void zero_f16_kernel(_Float16* p, long n) {
  for (long i = blockIdx.x * (long)blockDim.x + threadIdx.x; i < n;
       i += (long)gridDim.x * blockDim.x)
    p[i] = (_Float16)0.0f;
}

// Per-timestep: X^T (for adj GEMM), masks, and q = m2 ? no_prev : u_emb.
__global__ void buildx_kernel(const float* __restrict__ code_x,
                              const float* __restrict__ neighbors,
                              const float* __restrict__ divided,
                              const float* __restrict__ c_emb,
                              const float* __restrict__ n_emb,
                              const _Float16* __restrict__ u_h,
                              const _Float16* __restrict__ no_prev_h,
                              _Float16* __restrict__ Xt,
                              _Float16* __restrict__ q_h,
                              float* __restrict__ m1w, float* __restrict__ m23w, int t) {
  int i = blockIdx.x * blockDim.x + threadIdx.x;
  if (i >= kB * kN) return;
  int b = i / kN, n = i % kN;
  long src = ((long)b * kT + t) * kN + n;
  float c = code_x[src], nb = neighbors[src];
  m1w[i] = divided[src * 3 + 0];
  float m23 = divided[src * 3 + 1] + divided[src * 3 + 2];
  if (m23 > 1.0f) m23 = 1.0f;
  if (t == 0) m23 = 0.0f;                 // (t_pos > 0) factor
  m23w[i] = m23;
  const _Float16* qs = (divided[src * 3 + 1] > 0.0f) ? (no_prev_h + (long)i * kG)
                                                     : (u_h + (long)n * kG);
  for (int j = 0; j < kG; ++j) {
    q_h[(long)i * kG + j] = qs[j];
    float x = c * c_emb[n * kCS + j] + nb * n_emb[n * kCS + j];
    Xt[(long)b * kCS * kN + (long)j * kN + n] = (_Float16)x;
  }
}

// ci = c*(c_emb+agg); ni = n*(n_emb+agg)   (inputs to the Wg projection)
__global__ void buildci_kernel(const float* __restrict__ code_x,
                               const float* __restrict__ neighbors,
                               const float* __restrict__ c_emb,
                               const float* __restrict__ n_emb,
                               const float* __restrict__ agg,
                               _Float16* __restrict__ ci_h, _Float16* __restrict__ ni_h, int t) {
  int i = blockIdx.x * blockDim.x + threadIdx.x;
  if (i >= kB * kN) return;
  int b = i / kN, n = i % kN;
  long src = ((long)b * kT + t) * kN + n;
  float c = code_x[src], nb = neighbors[src];
  for (int j = 0; j < kCS; ++j) {
    float a = agg[(long)i * kCS + j];
    ci_h[(long)i * kCS + j] = (_Float16)(c * (c_emb[n * kCS + j] + a));
    ni_h[(long)i * kCS + j] = (_Float16)(nb * (n_emb[n * kCS + j] + a));
  }
}

// GRU gates (torch order r,z,n): g = (1-z)*tanh(ig + r*hg) + z*h
__global__ void gates_kernel(const float* __restrict__ gi, const float* __restrict__ gh,
                             const float* __restrict__ h, float* __restrict__ g, long total) {
  long i = blockIdx.x * (long)blockDim.x + threadIdx.x;
  if (i >= total) return;
  long row = i / kH;
  int j = (int)(i % kH);
  long base = row * 3 * kH;
  float r = sigmoidf_(gi[base + j] + gh[base + j]);
  float z = sigmoidf_(gi[base + kH + j] + gh[base + kH + j]);
  float gg = tanhf(gi[base + 2 * kH + j] + r * gh[base + 2 * kH + j]);
  g[i] = (1.0f - z) * gg + z * h[i];
}

// masked maxes over codes + new hidden state; captures outs[t] / out_last.
__global__ __launch_bounds__(128) void reduce_kernel(
    const float* __restrict__ g, const float* __restrict__ attn,
    const float* __restrict__ m1w, const float* __restrict__ m23w,
    float* __restrict__ h, _Float16* __restrict__ h_h,
    float* __restrict__ outs, float* __restrict__ out_last,
    const int* __restrict__ lens, int t) {
  int b = blockIdx.x, j = threadIdx.x;
  float mx1 = -1e30f, mx2 = -1e30f;
  int has1 = 0, has2 = 0;
  for (int n = 0; n < kN; ++n) {
    long r = (long)b * kN + n;
    float m1 = m1w[r], m23 = m23w[r];
    float gv = g[r * kH + j], av = attn[r * kH + j];
    if (m1 > 0.0f)  { has1 = 1; mx1 = fmaxf(mx1, gv); }
    if (m23 > 0.0f) { has2 = 1; mx2 = fmaxf(mx2, av); }
    float hn = m1 * gv + m23 * av;
    h[r * kH + j] = hn;
    h_h[r * kH + j] = (_Float16)hn;
  }
  float o = (has1 ? mx1 : 0.0f) + (has2 ? mx2 : 0.0f);
  outs[((long)t * kB + b) * kH + j] = o;
  if (t == lens[b] - 1) out_last[b * kH + j] = o;
}

__global__ void capture_last_kernel(const _Float16* __restrict__ no_h,
                                    const float* __restrict__ divided,
                                    _Float16* __restrict__ no_last_h,
                                    float* __restrict__ d_last,
                                    const int* __restrict__ lens, int t) {
  int i = blockIdx.x * blockDim.x + threadIdx.x;
  if (i >= kB * kN) return;
  int b = i / kN, n = i % kN;
  if (lens[b] - 1 != t) return;
  for (int j = 0; j < kG; ++j) no_last_h[(long)i * kG + j] = no_h[(long)i * kG + j];
  long src = ((long)b * kT + t) * kN + n;
  for (int j = 0; j < 3; ++j) d_last[(long)i * 3 + j] = divided[src * 3 + j];
}

// virtual last visit: co=sc broadcast, q = m2 ? no_last : su, h = out_last bcast
__global__ void build_final_kernel(const float* __restrict__ d_last,
                                   const _Float16* __restrict__ no_last_h,
                                   const _Float16* __restrict__ su_h,
                                   const _Float16* __restrict__ sc_h,
                                   const float* __restrict__ out_last,
                                   _Float16* __restrict__ q_h, _Float16* __restrict__ co_h,
                                   float* __restrict__ m1w, float* __restrict__ m23w,
                                   float* __restrict__ h, _Float16* __restrict__ h_h) {
  int i = blockIdx.x * blockDim.x + threadIdx.x;
  if (i >= kB * kN) return;
  int b = i / kN;
  float m2 = d_last[(long)i * 3 + 1];
  m1w[i] = d_last[(long)i * 3 + 0];
  float m23 = m2 + d_last[(long)i * 3 + 2];
  if (m23 > 1.0f) m23 = 1.0f;
  m23w[i] = m23;                                   // t_pos == 1, no suppression
  for (int j = 0; j < kG; ++j) {
    q_h[(long)i * kG + j] = (m2 > 0.0f) ? no_last_h[(long)i * kG + j] : su_h[j];
    co_h[(long)i * kG + j] = sc_h[j];
  }
  for (int j = 0; j < kH; ++j) {
    float v = out_last[b * kH + j];
    h[(long)i * kH + j] = v;
    h_h[(long)i * kH + j] = (_Float16)v;
  }
}

__global__ __launch_bounds__(128) void final_reduce_kernel(
    const float* __restrict__ g, const float* __restrict__ attn,
    const float* __restrict__ m1w, const float* __restrict__ m23w,
    float* __restrict__ visit) {
  int b = blockIdx.x, j = threadIdx.x;
  float mx1 = -1e30f, mx2 = -1e30f;
  int has1 = 0, has2 = 0;
  for (int n = 0; n < kN; ++n) {
    long r = (long)b * kN + n;
    if (m1w[r] > 0.0f)  { has1 = 1; mx1 = fmaxf(mx1, g[r * kH + j]); }
    if (m23w[r] > 0.0f) { has2 = 1; mx2 = fmaxf(mx2, attn[r * kH + j]); }
  }
  visit[b * kH + j] = (has1 ? mx1 : 0.0f) + (has2 ? mx2 : 0.0f);
}

// pooled = (sum_t outs*mask + visit)/lens ; logits = pooled @ Wc + bc
__global__ __launch_bounds__(256) void output_kernel(
    const float* __restrict__ outs, const float* __restrict__ visit,
    const int* __restrict__ lens, const float* __restrict__ Wc,
    const float* __restrict__ bc, float* __restrict__ out) {
  __shared__ float pooled[kH];
  int b = blockIdx.x, tid = threadIdx.x;
  int L = lens[b];
  if (tid < kH) {
    float s = visit[b * kH + tid];
    for (int tt = 0; tt < kT; ++tt)
      if (tt < L) s += outs[((long)tt * kB + b) * kH + tid];
    pooled[tid] = s / (float)L;
  }
  __syncthreads();
  for (int o = tid; o < kOUT; o += 256) {
    float acc = bc[o];
    for (int j = 0; j < kH; ++j) acc = fmaf(pooled[j], Wc[(long)j * kOUT + o], acc);
    out[(long)b * kOUT + o] = acc;
  }
}

// ---------------------------------------------------------------------------

extern "C" void kernel_launch(void* const* d_in, const int* in_sizes, int n_in,
                              void* d_out, int out_size, void* d_ws, size_t ws_size,
                              hipStream_t stream) {
  (void)in_sizes; (void)n_in; (void)out_size; (void)ws_size;
  const float* code_x    = (const float*)d_in[0];
  const float* divided   = (const float*)d_in[1];
  const float* neighbors = (const float*)d_in[2];
  const int*   lens      = (const int*)d_in[3];
  // d_in[4] intervals: unused by the NoTime path
  const float* adj  = (const float*)d_in[5];
  const float* cEmb = (const float*)d_in[6];
  const float* nEmb = (const float*)d_in[7];
  const float* uEmb = (const float*)d_in[8];
  const float* Wg   = (const float*)d_in[9];
  const float* bg   = (const float*)d_in[10];
  const float* Wih  = (const float*)d_in[11];
  const float* bih  = (const float*)d_in[12];
  const float* Whh  = (const float*)d_in[13];
  const float* bhh  = (const float*)d_in[14];
  const float* Wq   = (const float*)d_in[15];
  const float* bq   = (const float*)d_in[16];
  const float* Wk   = (const float*)d_in[17];
  const float* bk   = (const float*)d_in[18];
  const float* Wv   = (const float*)d_in[19];
  const float* bv   = (const float*)d_in[20];
  const float* Wc   = (const float*)d_in[21];
  const float* bc   = (const float*)d_in[22];
  const float* sc   = (const float*)d_in[23];
  const float* su   = (const float*)d_in[24];
  float* out = (float*)d_out;

  char* ws = (char*)d_ws;
  size_t off = 0;
  auto alloc = [&](size_t bytes) -> char* {
    char* p = ws + off;
    off = (off + bytes + 255) & ~(size_t)255;
    return p;
  };

  _Float16* adj_h  = (_Float16*)alloc((size_t)kN * kN * 2);
  _Float16* WgT    = (_Float16*)alloc((size_t)kG * kCS * 2);
  _Float16* WihT   = (_Float16*)alloc((size_t)3 * kH * kG * 2);
  _Float16* WhhT   = (_Float16*)alloc((size_t)3 * kH * kH * 2);
  _Float16* WqT    = (_Float16*)alloc((size_t)kA * kG * 2);
  _Float16* WkT    = (_Float16*)alloc((size_t)kA * kG * 2);
  _Float16* WvT    = (_Float16*)alloc((size_t)kH * kG * 2);
  _Float16* u_h    = (_Float16*)alloc((size_t)kN * kG * 2);
  _Float16* sc_h   = (_Float16*)alloc((size_t)kG * 2);
  _Float16* su_h   = (_Float16*)alloc((size_t)kG * 2);
  _Float16* Xt     = (_Float16*)alloc((size_t)kB * kCS * kN * 2);
  float*    agg    = (float*)   alloc((size_t)kB * kN * kCS * 4);
  _Float16* ci_h   = (_Float16*)alloc((size_t)kB * kN * kCS * 2);
  _Float16* ni_h   = (_Float16*)alloc((size_t)kB * kN * kCS * 2);
  _Float16* co_h   = (_Float16*)alloc((size_t)kB * kN * kG * 2);
  _Float16* noh0   = (_Float16*)alloc((size_t)kB * kN * kG * 2);
  _Float16* noh1   = (_Float16*)alloc((size_t)kB * kN * kG * 2);
  _Float16* q_h    = (_Float16*)alloc((size_t)kB * kN * kG * 2);
  float*    gi     = (float*)   alloc((size_t)kB * kN * 3 * kH * 4);
  float*    gh     = (float*)   alloc((size_t)kB * kN * 3 * kH * 4);
  float*    gbuf   = (float*)   alloc((size_t)kB * kN * kH * 4);
  float*    hbuf   = (float*)   alloc((size_t)kB * kN * kH * 4);
  _Float16* h_h    = (_Float16*)alloc((size_t)kB * kN * kH * 2);
  _Float16* Qh     = (_Float16*)alloc((size_t)kB * kN * kA * 2);
  _Float16* Kh     = (_Float16*)alloc((size_t)kB * kN * kA * 2);
  _Float16* Vt     = (_Float16*)alloc((size_t)kB * kH * kN * 2);
  float*    attnb  = (float*)   alloc((size_t)kB * kN * kH * 4);
  float*    m1w    = (float*)   alloc((size_t)kB * kN * 4);
  float*    m23w   = (float*)   alloc((size_t)kB * kN * 4);
  float*    outs   = (float*)   alloc((size_t)kT * kB * kH * 4);
  float*    outLast= (float*)   alloc((size_t)kB * kH * 4);
  _Float16* noLast = (_Float16*)alloc((size_t)kB * kN * kG * 2);
  float*    dLast  = (float*)   alloc((size_t)kB * kN * 3 * 4);
  float*    visit  = (float*)   alloc((size_t)kB * kH * 4);

  dim3 blk(256);
  auto gemm = [&](const _Float16* A, int lda, long sA,
                  const _Float16* Bt, int ldb, long sBt,
                  float* C, int ldc, long sC, _Float16* Ch,
                  _Float16* Ct, int ldct, long sCt,
                  const float* bias, int act, int K, int tilesM, int tilesN, int nbatch) {
    int total = nbatch * tilesM * tilesN;
    wmma_gemm_kernel<<<dim3((total + 7) / 8), blk, 0, stream>>>(
        A, lda, sA, Bt, ldb, sBt, C, ldc, sC, Ch, Ct, ldct, sCt,
        bias, act, K, tilesM, tilesN, total);
  };

  // ---- one-time prep: f16 conversions / K-major transposes, zero state ----
  cvt_f16_kernel<<<dim3(1024), blk, 0, stream>>>(adj_h, adj, (long)kN * kN);
  transpose_f16_kernel<<<dim3(64),  blk, 0, stream>>>(WgT,  Wg,  kCS, kG);
  transpose_f16_kernel<<<dim3(96),  blk, 0, stream>>>(WihT, Wih, kG, 3 * kH);
  transpose_f16_kernel<<<dim3(192), blk, 0, stream>>>(WhhT, Whh, kH, 3 * kH);
  transpose_f16_kernel<<<dim3(8),   blk, 0, stream>>>(WqT,  Wq,  kG, kA);
  transpose_f16_kernel<<<dim3(8),   blk, 0, stream>>>(WkT,  Wk,  kG, kA);
  transpose_f16_kernel<<<dim3(32),  blk, 0, stream>>>(WvT,  Wv,  kG, kH);
  cvt_f16_kernel<<<dim3(256), blk, 0, stream>>>(u_h, uEmb, (long)kN * kG);
  cvt_f16_kernel<<<dim3(1), blk, 0, stream>>>(sc_h, sc, kG);
  cvt_f16_kernel<<<dim3(1), blk, 0, stream>>>(su_h, su, kG);
  zero_f32_kernel<<<dim3(1024), blk, 0, stream>>>(hbuf, (long)kB * kN * kH);
  zero_f16_kernel<<<dim3(1024), blk, 0, stream>>>(h_h,  (long)kB * kN * kH);
  zero_f16_kernel<<<dim3(512),  blk, 0, stream>>>(noh0, (long)kB * kN * kG);
  zero_f16_kernel<<<dim3(512),  blk, 0, stream>>>(noh1, (long)kB * kN * kG);

  const int tBN = (kB * kN) / 16;   // 512 row tiles when batches are merged

  // ---------------------------- 16-step scan ------------------------------
  for (int t = 0; t < kT; ++t) {
    _Float16* noCur = (t & 1) ? noh1 : noh0;
    const _Float16* noPrev = (t & 1) ? noh0 : noh1;

    buildx_kernel<<<dim3(32), blk, 0, stream>>>(code_x, neighbors, divided, cEmb, nEmb,
                                                u_h, noPrev, Xt, q_h, m1w, m23w, t);
    // agg[b] = adj @ X[b]   (TDM-staged B panel, f16 WMMA, f32 accumulate)
    adj_gemm_tdm_kernel<<<dim3(kB * (kN / 128)), blk, 0, stream>>>(adj_h, Xt, agg);
    buildci_kernel<<<dim3(32), blk, 0, stream>>>(code_x, neighbors, cEmb, nEmb, agg, ci_h, ni_h, t);
    // co/no = leaky_relu(ci/ni @ Wg + bg)
    gemm(ci_h, kCS, 0, WgT, kCS, 0, nullptr, kG, 0, co_h, nullptr, 0, 0, bg, 1, kCS, tBN, kG / 16, 1);
    gemm(ni_h, kCS, 0, WgT, kCS, 0, nullptr, kG, 0, noCur, nullptr, 0, 0, bg, 1, kCS, tBN, kG / 16, 1);
    // GRU pre-activations
    gemm(co_h, kG, 0, WihT, kG, 0, gi, 3 * kH, 0, nullptr, nullptr, 0, 0, bih, 0, kG, tBN, (3 * kH) / 16, 1);
    gemm(h_h,  kH, 0, WhhT, kH, 0, gh, 3 * kH, 0, nullptr, nullptr, 0, 0, bhh, 0, kH, tBN, (3 * kH) / 16, 1);
    gates_kernel<<<dim3(4096), blk, 0, stream>>>(gi, gh, hbuf, gbuf, (long)kB * kN * kH);
    // attention projections
    gemm(q_h, kG, 0, WqT, kG, 0, nullptr, kA, 0, Qh, nullptr, 0, 0, bq, 0, kG, tBN, kA / 16, 1);
    gemm(q_h, kG, 0, WkT, kG, 0, nullptr, kA, 0, Kh, nullptr, 0, 0, bk, 0, kG, tBN, kA / 16, 1);
    gemm(co_h, kG, (long)kN * kG, WvT, kG, 0, nullptr, kH, 0, nullptr,
         Vt, kN, (long)kH * kN, bv, 0, kG, kN / 16, kH / 16, kB);   // V stored K-major
    attn_kernel<<<dim3(kB * (kN / 16) / 8), blk, 0, stream>>>(Qh, Kh, Vt, m23w, attnb);
    reduce_kernel<<<dim3(kB), dim3(kH), 0, stream>>>(gbuf, attnb, m1w, m23w, hbuf, h_h,
                                                     outs, outLast, lens, t);
    capture_last_kernel<<<dim3(32), blk, 0, stream>>>(noCur, divided, noLast, dLast, lens, t);
  }

  // -------------------------- virtual last visit --------------------------
  build_final_kernel<<<dim3(32), blk, 0, stream>>>(dLast, noLast, su_h, sc_h, outLast,
                                                   q_h, co_h, m1w, m23w, hbuf, h_h);
  gemm(co_h, kG, 0, WihT, kG, 0, gi, 3 * kH, 0, nullptr, nullptr, 0, 0, bih, 0, kG, tBN, (3 * kH) / 16, 1);
  gemm(h_h,  kH, 0, WhhT, kH, 0, gh, 3 * kH, 0, nullptr, nullptr, 0, 0, bhh, 0, kH, tBN, (3 * kH) / 16, 1);
  gates_kernel<<<dim3(4096), blk, 0, stream>>>(gi, gh, hbuf, gbuf, (long)kB * kN * kH);
  gemm(q_h, kG, 0, WqT, kG, 0, nullptr, kA, 0, Qh, nullptr, 0, 0, bq, 0, kG, tBN, kA / 16, 1);
  gemm(q_h, kG, 0, WkT, kG, 0, nullptr, kA, 0, Kh, nullptr, 0, 0, bk, 0, kG, tBN, kA / 16, 1);
  gemm(co_h, kG, (long)kN * kG, WvT, kG, 0, nullptr, kH, 0, nullptr,
       Vt, kN, (long)kH * kN, bv, 0, kG, kN / 16, kH / 16, kB);
  attn_kernel<<<dim3(kB * (kN / 16) / 8), blk, 0, stream>>>(Qh, Kh, Vt, m23w, attnb);
  final_reduce_kernel<<<dim3(kB), dim3(kH), 0, stream>>>(gbuf, attnb, m1w, m23w, visit);

  // ------------------------------- output ---------------------------------
  output_kernel<<<dim3(kB), blk, 0, stream>>>(outs, visit, lens, Wc, bc, out);
}
